// SparseRouter_5978594476067
// MI455X (gfx1250) — matrix operand order repaired and checked by reading, not compiled, not run on hardware
//
#include <hip/hip_runtime.h>
#include <hip/hip_bf16.h>
#include <math.h>

// ---------------- problem constants ----------------
#define HID   2048      // HIDDEN
#define FFN_  8192
#define BN    128       // BOTTLENECK
#define M_TOK 4096      // B*S
#define KSEL  819       // int(FFN * (1-0.9))
#define KPAD  832       // padded to multiple of 64 (13*64)

// ---------------- types for WMMA ----------------
typedef __bf16 bf16x8  __attribute__((ext_vector_type(8)));
typedef __bf16 bf16x16 __attribute__((ext_vector_type(16)));
typedef float  f32x8   __attribute__((ext_vector_type(8)));

__device__ __forceinline__ unsigned short f2bf(float x) {
    unsigned u = __float_as_uint(x);
    unsigned r = (u + 0x7FFFu + ((u >> 16) & 1u)) >> 16;   // round-to-nearest-even
    return (unsigned short)r;
}

// ---------------- workspace layout (bytes) ----------------
#define OFF_SCORES 0u                                   // 8192 * 4      = 32 KB
#define OFF_IDX    32768u                               // 832  * 4
#define OFF_HIDB   65536u                               // 4096*2048*2   = 16 MB
#define OFF_GWB    (OFF_HIDB + (size_t)M_TOK*HID*2)     // 832*2048*2
#define OFF_UWB    (OFF_GWB  + (size_t)KPAD*HID*2)
#define OFF_DWN    (OFF_UWB  + (size_t)KPAD*HID*2)      // 2048*832*2
#define OFF_ACT    (OFF_DWN  + (size_t)HID*KPAD*2)      // 4096*832*2

// =====================================================================
// Router: scores[0,0,:] only (that's all top_k consumes).
// =====================================================================
__global__ __launch_bounds__(256) void router_kernel(
    const float* __restrict__ hs,   // hidden_state, token (0,0) = first row
    const float* __restrict__ w1, const float* __restrict__ b1,
    const float* __restrict__ w2, const float* __restrict__ b2,
    float* __restrict__ scores)
{
    __shared__ float hbn[BN];
    const int tid = threadIdx.x;
    if (tid < BN) {
        float s = b1[tid];
        const float* wr = w1 + (size_t)tid * HID;
        #pragma unroll 8
        for (int h = 0; h < HID; ++h) s += hs[h] * wr[h];
        // exact gelu: 0.5*x*(1+erf(x/sqrt(2)))
        hbn[tid] = 0.5f * s * (1.0f + erff(s * 0.70710678118654752f));
    }
    __syncthreads();
    for (int f = tid; f < FFN_; f += 256) {
        float s = b2[f];
        const float* wr = w2 + (size_t)f * BN;
        #pragma unroll 8
        for (int c = 0; c < BN; ++c) s += hbn[c] * wr[c];
        scores[f] = s;
    }
}

// =====================================================================
// Top-K (K=819) via radix select on order-preserving uint keys, then a
// deterministic ascending-index emit (ties broken by lowest index, which
// matches jax.lax.top_k followed by sort).
// =====================================================================
__global__ __launch_bounds__(256) void topk_kernel(
    const float* __restrict__ scores, int* __restrict__ idx_out)
{
    __shared__ unsigned key[FFN_];       // 32 KB
    __shared__ unsigned s_cnt;
    __shared__ unsigned s_prefix;
    __shared__ int      s_kr;
    __shared__ int      s_gt[256], s_eq[256];
    __shared__ int      s_gt_off[257], s_eq_off[257];

    const int tid = threadIdx.x;
    for (int f = tid; f < FFN_; f += 256) {
        unsigned u = __float_as_uint(scores[f]);
        key[f] = (u & 0x80000000u) ? ~u : (u | 0x80000000u);  // monotone map
    }
    if (tid == 0) { s_prefix = 0u; s_kr = KSEL; }
    __syncthreads();

    for (int bit = 31; bit >= 0; --bit) {
        if (tid == 0) s_cnt = 0u;
        __syncthreads();
        const unsigned maskAbove = (bit == 0) ? 0xFFFFFFFFu : ~((1u << bit) - 1u);
        const unsigned test = s_prefix | (1u << bit);
        int local = 0;
        for (int f = tid; f < FFN_; f += 256)
            if ((key[f] & maskAbove) == test) ++local;
        atomicAdd(&s_cnt, (unsigned)local);
        __syncthreads();
        if (tid == 0) {
            if ((int)s_cnt >= s_kr) s_prefix = test;
            else                    s_kr -= (int)s_cnt;
        }
        __syncthreads();
    }
    const unsigned T = s_prefix;   // K-th largest key value

    // contiguous 32-index chunk per thread -> prefix sums -> ordered emit
    const int c0 = tid * (FFN_ / 256), c1 = c0 + (FFN_ / 256);
    int ngt = 0, neq = 0;
    for (int i = c0; i < c1; ++i) {
        if (key[i] > T) ++ngt; else if (key[i] == T) ++neq;
    }
    s_gt[tid] = ngt; s_eq[tid] = neq;
    __syncthreads();
    if (tid == 0) {
        int ag = 0, ae = 0;
        for (int t = 0; t < 256; ++t) {
            s_gt_off[t] = ag; ag += s_gt[t];
            s_eq_off[t] = ae; ae += s_eq[t];
        }
        s_gt_off[256] = ag; s_eq_off[256] = ae;
    }
    __syncthreads();
    const int quota  = KSEL - s_gt_off[256];            // >= 1 by construction
    int eqrank = s_eq_off[tid];
    int pos    = s_gt_off[tid] + min(s_eq_off[tid], quota);
    for (int i = c0; i < c1; ++i) {
        const unsigned k = key[i];
        if (k > T)        { idx_out[pos++] = i; }
        else if (k == T)  { if (eqrank < quota) idx_out[pos++] = i; ++eqrank; }
    }
}

// =====================================================================
// Converters / gathers (fp32 -> bf16, pad rows >= KSEL with zeros)
// =====================================================================
__global__ __launch_bounds__(256) void cvt_hidden_kernel(
    const float* __restrict__ hs, unsigned short* __restrict__ hidB, int total)
{
    int i = blockIdx.x * 256 + threadIdx.x;
    const int stride = gridDim.x * 256;
    for (; i < total; i += stride) hidB[i] = f2bf(hs[i]);
}

__global__ __launch_bounds__(256) void gather_gu_kernel(
    const float* __restrict__ gw, const float* __restrict__ uw,
    const int* __restrict__ idx,
    unsigned short* __restrict__ gwB, unsigned short* __restrict__ uwB)
{
    const int r = blockIdx.x;                    // 0..KPAD-1 (expert slot)
    const size_t dst = (size_t)r * HID;
    if (r < KSEL) {
        const int e = idx[r];
        const float* gs = gw + (size_t)e * HID;
        const float* us = uw + (size_t)e * HID;
        for (int h = threadIdx.x; h < HID; h += 256) {
            gwB[dst + h] = f2bf(gs[h]);
            uwB[dst + h] = f2bf(us[h]);
        }
    } else {
        for (int h = threadIdx.x; h < HID; h += 256) {
            gwB[dst + h] = 0; uwB[dst + h] = 0;
        }
    }
}

__global__ __launch_bounds__(256) void gather_down_kernel(
    const float* __restrict__ dw, const int* __restrict__ idx,
    unsigned short* __restrict__ dwN)
{
    const int h = blockIdx.x;                    // 0..HID-1
    const size_t dst = (size_t)h * KPAD;
    const float* src = dw + (size_t)h * FFN_;
    for (int k = threadIdx.x; k < KPAD; k += 256)
        dwN[dst + k] = (k < KSEL) ? f2bf(src[idx[k]]) : (unsigned short)0;
}

// =====================================================================
// WMMA fragment helpers (contiguous k = half*16 + s for A and B, not ISA table order; wave32)
//   A 16x32 bf16: lane=(half*16+m); chunks at K = half*16 and half*16+8
//   B (N-major rows, contiguous K): lane=(half*16+n); 32B at K = half*16
// =====================================================================
__device__ __forceinline__ bf16x16 load_fragA(const unsigned short* rowPtr,
                                              int k0, int half)
{
    bf16x8 a0 = *reinterpret_cast<const bf16x8*>(rowPtr + k0 + half * 16);
    bf16x8 a1 = *reinterpret_cast<const bf16x8*>(rowPtr + k0 + half * 16 + 8);
    bf16x16 a;
    #pragma unroll
    for (int i = 0; i < 8; ++i) { a[i] = a0[i]; a[i + 8] = a1[i]; }
    return a;
}

__device__ __forceinline__ bf16x16 load_fragB(const unsigned short* rowPtr,
                                              int k0, int half)
{
    return *reinterpret_cast<const bf16x16*>(rowPtr + k0 + half * 16);
}

// =====================================================================
// Fused gate/up GEMM + SiLU.  A = hidB (M=4096, K=2048, row-major bf16),
// B = gwB/uwB (N-major, KPAD rows x HID).  Each wave: 16(M) x 64(N) tile,
// gate and up accumulated together (A fragments shared) -> act bf16.
// Ping-pong pipeline, manually unrolled x2: buffers live in fixed
// registers (no rotation copies), loads issued one full k-step ahead.
// grid = (M/128, KPAD/64), block = 256 (8 waves).
// =====================================================================
__global__ __launch_bounds__(256) void gateup_kernel(
    const unsigned short* __restrict__ hidB,
    const unsigned short* __restrict__ gwB,
    const unsigned short* __restrict__ uwB,
    unsigned short* __restrict__ act)
{
    const int lane = threadIdx.x & 31;
    const int wave = threadIdx.x >> 5;
    const int lm   = lane & 15;
    const int half = lane >> 4;
    const int mBase = blockIdx.x * 128 + wave * 16;
    const int nBase = blockIdx.y * 64;

    f32x8 accg[4], accu[4];
    const f32x8 zero = {};
    #pragma unroll
    for (int j = 0; j < 4; ++j) { accg[j] = zero; accu[j] = zero; }

    const unsigned short* pA = hidB + (size_t)(mBase + lm) * HID;
    const unsigned short* pG[4];
    const unsigned short* pU[4];
    #pragma unroll
    for (int j = 0; j < 4; ++j) {
        const size_t row = (size_t)(nBase + j * 16 + lm);
        pG[j] = gwB + row * HID;
        pU[j] = uwB + row * HID;
    }

    bf16x16 a0, a1, g0[4], u0[4], g1[4], u1[4];

    // prologue: step 0 into buffer 0
    a0 = load_fragA(pA, 0, half);
    #pragma unroll
    for (int j = 0; j < 4; ++j) {
        g0[j] = load_fragB(pG[j], 0, half);
        u0[j] = load_fragB(pU[j], 0, half);
    }

    // main loop: 2 k-steps per iteration, no buffer copies.
    // STEPS = HID/32 = 64 (even); iterations = (STEPS-2)/2 = 31.
    for (int k0 = 0; k0 < HID - 64; k0 += 64) {
        __builtin_prefetch(pA + k0 + 512, 0, 3);  // stream A into GL2
        // load buffer1 (step k0+32), consume buffer0 (step k0)
        a1 = load_fragA(pA, k0 + 32, half);
        #pragma unroll
        for (int j = 0; j < 4; ++j) {
            g1[j] = load_fragB(pG[j], k0 + 32, half);
            u1[j] = load_fragB(pU[j], k0 + 32, half);
        }
        #pragma unroll
        for (int j = 0; j < 4; ++j) {
            accg[j] = __builtin_amdgcn_wmma_f32_16x16x32_bf16(
                false, a0, false, g0[j], (short)0, accg[j], false, false);
            accu[j] = __builtin_amdgcn_wmma_f32_16x16x32_bf16(
                false, a0, false, u0[j], (short)0, accu[j], false, false);
        }
        // load buffer0 (step k0+64), consume buffer1 (step k0+32)
        a0 = load_fragA(pA, k0 + 64, half);
        #pragma unroll
        for (int j = 0; j < 4; ++j) {
            g0[j] = load_fragB(pG[j], k0 + 64, half);
            u0[j] = load_fragB(pU[j], k0 + 64, half);
        }
        #pragma unroll
        for (int j = 0; j < 4; ++j) {
            accg[j] = __builtin_amdgcn_wmma_f32_16x16x32_bf16(
                false, a1, false, g1[j], (short)0, accg[j], false, false);
            accu[j] = __builtin_amdgcn_wmma_f32_16x16x32_bf16(
                false, a1, false, u1[j], (short)0, accu[j], false, false);
        }
    }
    // tail: buffer0 holds step HID-64; load+consume last step HID-32
    a1 = load_fragA(pA, HID - 32, half);
    #pragma unroll
    for (int j = 0; j < 4; ++j) {
        g1[j] = load_fragB(pG[j], HID - 32, half);
        u1[j] = load_fragB(pU[j], HID - 32, half);
    }
    #pragma unroll
    for (int j = 0; j < 4; ++j) {
        accg[j] = __builtin_amdgcn_wmma_f32_16x16x32_bf16(
            false, a0, false, g0[j], (short)0, accg[j], false, false);
        accu[j] = __builtin_amdgcn_wmma_f32_16x16x32_bf16(
            false, a0, false, u0[j], (short)0, accu[j], false, false);
    }
    #pragma unroll
    for (int j = 0; j < 4; ++j) {
        accg[j] = __builtin_amdgcn_wmma_f32_16x16x32_bf16(
            false, a1, false, g1[j], (short)0, accg[j], false, false);
        accu[j] = __builtin_amdgcn_wmma_f32_16x16x32_bf16(
            false, a1, false, u1[j], (short)0, accu[j], false, false);
    }

    // C/D layout: lane(n + 16*(m/8)), vgpr v = m%8
    #pragma unroll
    for (int j = 0; j < 4; ++j) {
        const int n = nBase + j * 16 + lm;
        #pragma unroll
        for (int v = 0; v < 8; ++v) {
            const int m = mBase + half * 8 + v;
            const float g = accg[j][v];
            const float u = accu[j][v];
            const float sig = 1.0f / (1.0f + __expf(-g));
            act[(size_t)m * KPAD + n] = f2bf(g * sig * u);
        }
    }
}

// =====================================================================
// Down GEMM: A = act (M=4096 x K=832 bf16), B = dwN (N-major, HID x KPAD),
// D = out f32 (4096 x 2048).  Same ping-pong pipeline (KPAD/32 = 26 even).
// grid = (M/128, HID/64), block = 256.
// =====================================================================
__global__ __launch_bounds__(256) void down_kernel(
    const unsigned short* __restrict__ act,
    const unsigned short* __restrict__ dwN,
    float* __restrict__ out)
{
    const int lane = threadIdx.x & 31;
    const int wave = threadIdx.x >> 5;
    const int lm   = lane & 15;
    const int half = lane >> 4;
    const int mBase = blockIdx.x * 128 + wave * 16;
    const int nBase = blockIdx.y * 64;

    f32x8 acc[4];
    const f32x8 zero = {};
    #pragma unroll
    for (int j = 0; j < 4; ++j) acc[j] = zero;

    const unsigned short* pA = act + (size_t)(mBase + lm) * KPAD;
    const unsigned short* pB[4];
    #pragma unroll
    for (int j = 0; j < 4; ++j)
        pB[j] = dwN + (size_t)(nBase + j * 16 + lm) * KPAD;

    bf16x16 a0, a1, b0[4], b1[4];

    a0 = load_fragA(pA, 0, half);
    #pragma unroll
    for (int j = 0; j < 4; ++j) b0[j] = load_fragB(pB[j], 0, half);

    for (int k0 = 0; k0 < KPAD - 64; k0 += 64) {
        __builtin_prefetch(pA + k0 + 512, 0, 3);
        a1 = load_fragA(pA, k0 + 32, half);
        #pragma unroll
        for (int j = 0; j < 4; ++j) b1[j] = load_fragB(pB[j], k0 + 32, half);
        #pragma unroll
        for (int j = 0; j < 4; ++j)
            acc[j] = __builtin_amdgcn_wmma_f32_16x16x32_bf16(
                false, a0, false, b0[j], (short)0, acc[j], false, false);
        a0 = load_fragA(pA, k0 + 64, half);
        #pragma unroll
        for (int j = 0; j < 4; ++j) b0[j] = load_fragB(pB[j], k0 + 64, half);
        #pragma unroll
        for (int j = 0; j < 4; ++j)
            acc[j] = __builtin_amdgcn_wmma_f32_16x16x32_bf16(
                false, a1, false, b1[j], (short)0, acc[j], false, false);
    }
    // tail: buffer0 holds step KPAD-64; last step KPAD-32 via buffer1
    a1 = load_fragA(pA, KPAD - 32, half);
    #pragma unroll
    for (int j = 0; j < 4; ++j) b1[j] = load_fragB(pB[j], KPAD - 32, half);
    #pragma unroll
    for (int j = 0; j < 4; ++j)
        acc[j] = __builtin_amdgcn_wmma_f32_16x16x32_bf16(
            false, a0, false, b0[j], (short)0, acc[j], false, false);
    #pragma unroll
    for (int j = 0; j < 4; ++j)
        acc[j] = __builtin_amdgcn_wmma_f32_16x16x32_bf16(
            false, a1, false, b1[j], (short)0, acc[j], false, false);

    #pragma unroll
    for (int j = 0; j < 4; ++j) {
        const int n = nBase + j * 16 + lm;
        #pragma unroll
        for (int v = 0; v < 8; ++v) {
            const int m = mBase + half * 8 + v;
            out[(size_t)m * HID + n] = acc[j][v];
        }
    }
}

// =====================================================================
// Launch
// =====================================================================
extern "C" void kernel_launch(void* const* d_in, const int* in_sizes, int n_in,
                              void* d_out, int out_size, void* d_ws, size_t ws_size,
                              hipStream_t stream)
{
    (void)in_sizes; (void)n_in; (void)out_size; (void)ws_size;
    const float* hs     = (const float*)d_in[0];  // (2,2048,2048)
    const float* r_w1   = (const float*)d_in[1];  // (128,2048)
    const float* r_b1   = (const float*)d_in[2];  // (128,)
    const float* r_w2   = (const float*)d_in[3];  // (8192,128)
    const float* r_b2   = (const float*)d_in[4];  // (8192,)
    const float* gate_w = (const float*)d_in[5];  // (8192,2048)
    const float* up_w   = (const float*)d_in[6];  // (8192,2048)
    const float* down_w = (const float*)d_in[7];  // (2048,8192)
    float* out = (float*)d_out;                   // (2,2048,2048) f32

    char* ws = (char*)d_ws;
    float*          scores = (float*)         (ws + OFF_SCORES);
    int*            idx    = (int*)           (ws + OFF_IDX);
    unsigned short* hidB   = (unsigned short*)(ws + OFF_HIDB);
    unsigned short* gwB    = (unsigned short*)(ws + OFF_GWB);
    unsigned short* uwB    = (unsigned short*)(ws + OFF_UWB);
    unsigned short* dwN    = (unsigned short*)(ws + OFF_DWN);
    unsigned short* act    = (unsigned short*)(ws + OFF_ACT);

    // 1) router scores for token (0,0) only
    router_kernel<<<1, 256, 0, stream>>>(hs, r_w1, r_b1, r_w2, r_b2, scores);
    // 2) top-819, sorted ascending
    topk_kernel<<<1, 256, 0, stream>>>(scores, idx);
    // 3) precision conversion + gathers (stream-ordered after idx)
    cvt_hidden_kernel<<<4096, 256, 0, stream>>>(hs, hidB, M_TOK * HID);
    gather_gu_kernel<<<KPAD, 256, 0, stream>>>(gate_w, up_w, idx, gwB, uwB);
    gather_down_kernel<<<HID, 256, 0, stream>>>(down_w, idx, dwN);
    // 4) fused gate/up GEMM + SiLU  (M=4096, N=832, K=2048)
    gateup_kernel<<<dim3(M_TOK / 128, KPAD / 64), 256, 0, stream>>>(hidB, gwB, uwB, act);
    // 5) down GEMM (M=4096, N=2048, K=832) -> f32 output
    down_kernel<<<dim3(M_TOK / 128, HID / 64), 256, 0, stream>>>(act, dwN, out);
}
